// meta_learning_2834678415425
// MI455X (gfx1250) — matrix operand order repaired and checked by reading, not compiled, not run on hardware
//
#include <hip/hip_runtime.h>
#include <math.h>

// ---------------------------------------------------------------------------
// Meta-learning (neural-process / GP head) pipeline for MI455X (gfx1250).
// Dominant GEMMs (Hk@W_K2 and Kq@Ks^T, ~13 GFLOP) run on
// v_wmma_f32_16x16x32_bf16 (wave32, 16x16 tiles, f32 accumulate).
// Layer-1 of the M/K MLPs is collapsed to 2 FMAs/element using the constant-Z
// structure of the concatenated inputs. pinv solve done via CG (minimal-norm
// solution of the SPD kernel system) in a single persistent workgroup.
// ---------------------------------------------------------------------------

#define DH 256
#define NS 512
#define NQ 32768

typedef __attribute__((ext_vector_type(16))) __bf16 v16bf;
typedef __attribute__((ext_vector_type(8)))  float  v8f;

union FragCast {
  uint4 q[2];
  v16bf v;
};

// round-to-nearest-even f32 -> bf16 (bit pattern)
static __device__ __forceinline__ unsigned short f2bf(float x) {
  unsigned int u = __float_as_uint(x);
  unsigned int r = u + 0x7FFFu + ((u >> 16) & 1u);
  return (unsigned short)(r >> 16);
}
static __device__ __forceinline__ float bf2f(unsigned short h) {
  return __uint_as_float(((unsigned int)h) << 16);
}

// Load a 16-element bf16 fragment from two 16-byte chunks (works for LDS
// and global pointers; lowers to ds_read_b128 / global_load_b128).
static __device__ __forceinline__ v16bf load_frag(const unsigned short* p0,
                                                  const unsigned short* p1) {
  FragCast f;
  f.q[0] = *(const uint4*)p0;
  f.q[1] = *(const uint4*)p1;
  return f.v;
}

static __device__ __forceinline__ v8f wmma_bf16(v16bf a, v16bf b, v8f c) {
  // (neg_a, A, neg_b, B, c_mod, C, reuse_a, reuse_b)
  return __builtin_amdgcn_wmma_f32_16x16x32_bf16(false, a, false, b,
                                                 (short)0, c, false, false);
}

// ---------------------------------------------------------------------------
// K1: h1mean = mean_i relu(support_i @ W_Z1 + b_Z1)   (mean commutes with L2)
// ---------------------------------------------------------------------------
__global__ __launch_bounds__(256) void k_embed(const float* __restrict__ support,
                                               const float* __restrict__ WZ1,
                                               const float* __restrict__ bZ1,
                                               float* __restrict__ h1mean) {
  __shared__ float sup[NS * 3];
  int t = threadIdx.x;
  for (int idx = t; idx < NS * 3; idx += 256) sup[idx] = support[idx];
  __syncthreads();
  float w0 = WZ1[t], w1 = WZ1[DH + t], w2 = WZ1[2 * DH + t], bb = bZ1[t];
  float acc = 0.f;
  for (int i = 0; i < NS; ++i) {
    float h = fmaf(sup[3 * i], w0, fmaf(sup[3 * i + 1], w1, fmaf(sup[3 * i + 2], w2, bb)));
    acc += fmaxf(h, 0.f);
  }
  h1mean[t] = acc * (1.0f / NS);
}

// ---------------------------------------------------------------------------
// K2: Z = h1mean @ W_Z2 + b_Z2
// ---------------------------------------------------------------------------
__global__ __launch_bounds__(256) void k_z(const float* __restrict__ h1mean,
                                           const float* __restrict__ WZ2,
                                           const float* __restrict__ bZ2,
                                           float* __restrict__ Z) {
  __shared__ float hm[DH];
  int t = threadIdx.x;
  hm[t] = h1mean[t];
  __syncthreads();
  float acc = bZ2[t];
  for (int k = 0; k < DH; ++k) acc = fmaf(hm[k], WZ2[k * DH + t], acc);
  Z[t] = acc;
}

// ---------------------------------------------------------------------------
// K3: constant-Z collapse vectors cM, cK; scalar b = MLP_B(Z)
// ---------------------------------------------------------------------------
__global__ __launch_bounds__(256) void k_context(const float* __restrict__ Zp,
                                                 const float* __restrict__ WM1,
                                                 const float* __restrict__ bM1,
                                                 const float* __restrict__ WK1,
                                                 const float* __restrict__ bK1,
                                                 const float* __restrict__ WB1,
                                                 const float* __restrict__ bB1,
                                                 const float* __restrict__ WB2,
                                                 const float* __restrict__ bB2,
                                                 float* __restrict__ cM,
                                                 float* __restrict__ cK,
                                                 float* __restrict__ bscal) {
  __shared__ float z[DH];
  __shared__ float red[DH];
  int t = threadIdx.x;
  z[t] = Zp[t];
  __syncthreads();
  float am = bM1[t], ak = bK1[t], ab = bB1[t];
  for (int k = 0; k < DH; ++k) {
    float zk = z[k];
    am = fmaf(zk, WM1[(2 + k) * DH + t], am);
    ak = fmaf(zk, WK1[(2 + k) * DH + t], ak);
    ab = fmaf(zk, WB1[k * DH + t], ab);
  }
  cM[t] = am;
  cK[t] = ak;
  float bh = fmaxf(ab, 0.f);
  red[t] = bh * WB2[t];
  __syncthreads();
  for (int s = 128; s > 0; s >>= 1) {
    if (t < s) red[t] += red[t + s];
    __syncthreads();
  }
  if (t == 0) bscal[0] = red[0] + bB2[0];
}

// ---------------------------------------------------------------------------
// K4: pack W_K2 (256x256 f32) into WMMA B-fragment layout (bf16), zero sn.
// B layout: [kt(8)][nt(16)][lane(32)][e(16)], k = kt*32 + (e&7) + 16*(e>=8) + 8*(lane>=16),
//           n = nt*16 + (lane&15)
// ---------------------------------------------------------------------------
__global__ __launch_bounds__(256) void k_pack_wk2(const float* __restrict__ WK2,
                                                  unsigned short* __restrict__ WK2p,
                                                  float* __restrict__ sn) {
  int flat = blockIdx.x * 256 + threadIdx.x;  // 65536 elements
  int e = flat & 15;
  int L = (flat >> 4) & 31;
  int nt = (flat >> 9) & 15;
  int kt = flat >> 13;
  int k = kt * 32 + (e & 7) + ((e & 8) ? 16 : 0) + ((L < 16) ? 0 : 8);
  int n = nt * 16 + (L & 15);
  WK2p[flat] = f2bf(WK2[k * DH + n]);
  if (flat < NS) sn[flat] = 0.f;
}

// ---------------------------------------------------------------------------
// K5: support path. Per 32-row tile: rhs = sy - mean-head; Hk -> WMMA GEMM1
// with packed W_K2 -> Ks (row-major bf16 + packed-B bf16) + norms sn.
// ---------------------------------------------------------------------------
__global__ __launch_bounds__(256) void k_support(const float* __restrict__ support,
                                                 const float* __restrict__ WM1,
                                                 const float* __restrict__ WM2,
                                                 const float* __restrict__ bM2,
                                                 const float* __restrict__ WK1,
                                                 const float* __restrict__ bK2v,
                                                 const float* __restrict__ cM,
                                                 const float* __restrict__ cK,
                                                 const unsigned short* __restrict__ WK2p,
                                                 unsigned short* __restrict__ KsRM,
                                                 unsigned short* __restrict__ KsP,
                                                 float* __restrict__ sn,
                                                 float* __restrict__ rhs) {
  __shared__ __align__(16) unsigned short hk[32 * DH];
  __shared__ float sxl0[32], sxl1[32], syl[32];
  __shared__ float red[256];
  int t = threadIdx.x;
  int r0 = blockIdx.x * 32;
  if (t < 32) {
    sxl0[t] = support[3 * (r0 + t)];
    sxl1[t] = support[3 * (r0 + t) + 1];
    syl[t]  = support[3 * (r0 + t) + 2];
  }
  __syncthreads();
  int i = t & 31, part = t >> 5;
  // mean head: support_mean, rhs = sy - mean
  {
    float x0 = sxl0[i], x1 = sxl1[i], acc = 0.f;
    for (int h = part * 32; h < part * 32 + 32; ++h) {
      float hv = fmaxf(fmaf(x0, WM1[h], fmaf(x1, WM1[DH + h], cM[h])), 0.f);
      acc = fmaf(hv, WM2[h], acc);
    }
    red[t] = acc;
    __syncthreads();
    if (part < 4) red[t] += red[t + 128];
    __syncthreads();
    if (part < 2) red[t] += red[t + 64];
    __syncthreads();
    if (part == 0) rhs[r0 + i] = syl[i] - (red[t] + red[t + 32] + bM2[0]);
  }
  __syncthreads();
  // Hk tile (bf16, row-major in LDS)
  for (int c = 0; c < 32; ++c) {
    int flat = t * 32 + c;
    int row = flat >> 8, h = flat & 255;
    float hv = fmaxf(fmaf(sxl0[row], WK1[h], fmaf(sxl1[row], WK1[DH + h], cK[h])), 0.f);
    hk[flat] = f2bf(hv);
  }
  __syncthreads();
  // GEMM1: Ks = Hk @ W_K2 + b_K2  (WMMA bf16)
  int lane = t & 31, w = t >> 5;
  int mt = w & 1;
  int laneLow = (lane < 16);
  int klo = laneLow ? 0 : 8;
  int mrb = mt * 16 + (laneLow ? 0 : 8);
  int am = mt * 16 + (lane & 15);
  float qnp[8] = {0.f, 0.f, 0.f, 0.f, 0.f, 0.f, 0.f, 0.f};
  for (int j = 0; j < 4; ++j) {
    int nt = (w >> 1) * 4 + j;
    v8f acc = {};
#pragma unroll
    for (int kb = 0; kb < 8; ++kb) {
      const unsigned short* ap = &hk[am * DH + kb * 32 + klo];
      v16bf A = load_frag(ap, ap + 16);
      const unsigned short* bp = &WK2p[(((kb * 16 + nt) * 32) + lane) * 16];
      v16bf B = load_frag(bp, bp + 8);
      acc = wmma_bf16(A, B, acc);
    }
    int f = nt * 16 + (lane & 15);
    float bk2 = bK2v[f];
    int kt2 = f >> 5, rem = f & 31;
    int e2 = (rem & 7) | ((rem >> 1) & 8);
    int lgrp = (rem & 8) ? 16 : 0;
#pragma unroll
    for (int r8 = 0; r8 < 8; ++r8) {
      int srow = r0 + mrb + r8;
      unsigned short bv = f2bf(acc[r8] + bk2);
      KsRM[srow * DH + f] = bv;
      int nt2 = (srow >> 4) & 31;
      int lane2 = lgrp + (srow & 15);
      KsP[(((kt2 * 32 + nt2) * 32) + lane2) * 16 + e2] = bv;
      float fv = bf2f(bv);
      qnp[r8] += fv * fv;
    }
  }
#pragma unroll
  for (int r8 = 0; r8 < 8; ++r8) atomicAdd(&sn[r0 + mrb + r8], qnp[r8]);
}

// ---------------------------------------------------------------------------
// K6: self_k = exp(-pdist(Ks,Ks)) + b*delta  (WMMA Gram, one 16x16 tile/wave)
// ---------------------------------------------------------------------------
__global__ __launch_bounds__(256) void k_selfk(const unsigned short* __restrict__ KsRM,
                                               const unsigned short* __restrict__ KsP,
                                               const float* __restrict__ sn,
                                               const float* __restrict__ support,
                                               const float* __restrict__ bscal,
                                               float* __restrict__ selfk) {
  int t = threadIdx.x, lane = t & 31, w = t >> 5;
  int wg = blockIdx.x * 8 + w;           // 0..1023
  int mt = wg >> 5, nt = wg & 31;
  int laneLow = (lane < 16);
  int klo = laneLow ? 0 : 8;
  int am = mt * 16 + (lane & 15);
  v8f acc = {};
#pragma unroll
  for (int kb = 0; kb < 8; ++kb) {
    const unsigned short* ap = &KsRM[am * DH + kb * 32 + klo];
    v16bf A = load_frag(ap, ap + 16);
    const unsigned short* bp = &KsP[(((kb * 32 + nt) * 32) + lane) * 16];
    v16bf B = load_frag(bp, bp + 8);
    acc = wmma_bf16(A, B, acc);
  }
  int n = nt * 16 + (lane & 15);
  float s0 = support[3 * n], s1 = support[3 * n + 1];
  float snn = sn[n], bcv = bscal[0];
#pragma unroll
  for (int r = 0; r < 8; ++r) {
    int m = mt * 16 + r + (laneLow ? 0 : 8);
    float d2 = sn[m] + snn - 2.f * acc[r];
    float v = expf(-sqrtf(fmaxf(d2, 0.f)));
    if (support[3 * m] == s0 && support[3 * m + 1] == s1) v += bcv;
    selfk[m * NS + n] = v;
  }
}

// ---------------------------------------------------------------------------
// K7: CG solve self_k x = rhs (SPD; CG -> minimal-norm / pinv solution).
// Single persistent workgroup, A L2-hot, vectors in LDS.
// ---------------------------------------------------------------------------
static __device__ __forceinline__ float reduce1024(float v, float* red, int t) {
  red[t] = v;
  __syncthreads();
  for (int s = 512; s > 0; s >>= 1) {
    if (t < s) red[t] += red[t + s];
    __syncthreads();
  }
  float r = red[0];
  __syncthreads();
  return r;
}

__global__ __launch_bounds__(1024) void k_cg(const float* __restrict__ A,
                                             const float* __restrict__ rhsv,
                                             float* __restrict__ xout) {
  __shared__ float p[NS], rv[NS], xv[NS], Ap[NS];
  __shared__ float red[1024];
  int t = threadIdx.x;
  if (t < NS) {
    xv[t] = 0.f;
    rv[t] = rhsv[t];
    p[t]  = rhsv[t];
  }
  __syncthreads();
  float rr = reduce1024((t < NS) ? rv[t] * rv[t] : 0.f, red, t);
  float rr0 = rr;
  for (int it = 0; it < 256; ++it) {
    int row = t >> 1, half = t & 1;
    const float* ar = A + row * NS + half * 256;
    const float* pr = p + half * 256;
    float a = 0.f;
    for (int c = 0; c < 256; ++c) a = fmaf(ar[c], pr[c], a);
    red[t] = a;
    __syncthreads();
    if (half == 0) Ap[row] = red[t] + red[t + 1];
    __syncthreads();
    float pAp = reduce1024((t < NS) ? p[t] * Ap[t] : 0.f, red, t);
    if (!(pAp > 1e-30f * rr0) || rr <= 1e-13f * rr0) break;  // uniform exit
    float alpha = rr / pAp;
    if (t < NS) {
      xv[t] = fmaf(alpha, p[t], xv[t]);
      rv[t] = fmaf(-alpha, Ap[t], rv[t]);
    }
    __syncthreads();
    float rr2 = reduce1024((t < NS) ? rv[t] * rv[t] : 0.f, red, t);
    float beta = rr2 / rr;
    if (t < NS) p[t] = fmaf(beta, p[t], rv[t]);
    rr = rr2;
    __syncthreads();
  }
  if (t < NS) xout[t] = xv[t];
}

// ---------------------------------------------------------------------------
// K8: fused query pipeline per 32-row tile:
//  mean head -> Hk -> WMMA GEMM1 (Kq) -> WMMA GEMM2 (Kq@Ks^T) -> exp/delta ->
//  y = qm + mix_k @ solve_vec
// ---------------------------------------------------------------------------
__global__ __launch_bounds__(256) void k_query(const float* __restrict__ query,
                                               const float* __restrict__ support,
                                               const float* __restrict__ WM1,
                                               const float* __restrict__ WM2,
                                               const float* __restrict__ bM2,
                                               const float* __restrict__ WK1,
                                               const float* __restrict__ bK2v,
                                               const float* __restrict__ cM,
                                               const float* __restrict__ cK,
                                               const unsigned short* __restrict__ WK2p,
                                               const unsigned short* __restrict__ KsP,
                                               const float* __restrict__ sn,
                                               const float* __restrict__ sv,
                                               const float* __restrict__ bscal,
                                               float* __restrict__ out) {
  __shared__ __align__(16) unsigned short hk[32 * DH];
  __shared__ __align__(16) unsigned short kq[32 * DH];
  __shared__ float sx0[NS], sx1[NS], svl[NS], snl[NS];
  __shared__ float qx0[32], qx1[32], qn[32], qm[32], yacc[32];
  __shared__ float red[256];

  int t = threadIdx.x;
  int q0 = blockIdx.x * 32;
  for (int idx = t; idx < NS; idx += 256) {
    sx0[idx] = support[3 * idx];
    sx1[idx] = support[3 * idx + 1];
    svl[idx] = sv[idx];
    snl[idx] = sn[idx];
  }
  if (t < 32) {
    qx0[t] = query[2 * (q0 + t)];
    qx1[t] = query[2 * (q0 + t) + 1];
    qn[t] = 0.f;
    yacc[t] = 0.f;
  }
  __syncthreads();

  // Phase A: query_mean head
  int i = t & 31, part = t >> 5;
  {
    float x0 = qx0[i], x1 = qx1[i], acc = 0.f;
    for (int h = part * 32; h < part * 32 + 32; ++h) {
      float hv = fmaxf(fmaf(x0, WM1[h], fmaf(x1, WM1[DH + h], cM[h])), 0.f);
      acc = fmaf(hv, WM2[h], acc);
    }
    red[t] = acc;
    __syncthreads();
    if (part < 4) red[t] += red[t + 128];
    __syncthreads();
    if (part < 2) red[t] += red[t + 64];
    __syncthreads();
    if (part == 0) qm[i] = red[t] + red[t + 32] + bM2[0];
  }
  __syncthreads();

  // Phase B: Hk tile (bf16 in LDS)
  for (int c = 0; c < 32; ++c) {
    int flat = t * 32 + c;
    int row = flat >> 8, h = flat & 255;
    float hv = fmaxf(fmaf(qx0[row], WK1[h], fmaf(qx1[row], WK1[DH + h], cK[h])), 0.f);
    hk[flat] = f2bf(hv);
  }
  __syncthreads();

  // Phase C: GEMM1  Kq = Hk @ W_K2 + b_K2  (WMMA bf16), + per-row norms
  int lane = t & 31, w = t >> 5;
  int mt = w & 1;
  int laneLow = (lane < 16);
  int klo = laneLow ? 0 : 8;
  int mrb = mt * 16 + (laneLow ? 0 : 8);
  int am = mt * 16 + (lane & 15);
  float qnp[8] = {0.f, 0.f, 0.f, 0.f, 0.f, 0.f, 0.f, 0.f};
  for (int j = 0; j < 4; ++j) {
    int nt = (w >> 1) * 4 + j;
    v8f acc = {};
#pragma unroll
    for (int kb = 0; kb < 8; ++kb) {
      const unsigned short* ap = &hk[am * DH + kb * 32 + klo];
      v16bf A = load_frag(ap, ap + 16);
      const unsigned short* bp = &WK2p[(((kb * 16 + nt) * 32) + lane) * 16];
      v16bf B = load_frag(bp, bp + 8);
      acc = wmma_bf16(A, B, acc);
    }
    int f = nt * 16 + (lane & 15);
    float bk2 = bK2v[f];
#pragma unroll
    for (int r8 = 0; r8 < 8; ++r8) {
      int m = mrb + r8;
      unsigned short bv = f2bf(acc[r8] + bk2);
      kq[m * DH + f] = bv;
      float fv = bf2f(bv);
      qnp[r8] += fv * fv;
    }
  }
#pragma unroll
  for (int r8 = 0; r8 < 8; ++r8) atomicAdd(&qn[mrb + r8], qnp[r8]);
  __syncthreads();

  // Phase D: GEMM2  dot = Kq @ Ks^T, fused exp(-dist)+b*delta, dot solve_vec
  float bcv = bscal[0];
  float yp[8] = {0.f, 0.f, 0.f, 0.f, 0.f, 0.f, 0.f, 0.f};
  for (int j = 0; j < 8; ++j) {
    int nt = (w >> 1) * 8 + j;
    v8f acc = {};
#pragma unroll
    for (int kb = 0; kb < 8; ++kb) {
      const unsigned short* ap = &kq[am * DH + kb * 32 + klo];
      v16bf A = load_frag(ap, ap + 16);
      const unsigned short* bp = &KsP[(((kb * 32 + nt) * 32) + lane) * 16];
      v16bf B = load_frag(bp, bp + 8);
      acc = wmma_bf16(A, B, acc);
    }
    int n = nt * 16 + (lane & 15);
    float s0 = sx0[n], s1 = sx1[n], svn = svl[n], snn = snl[n];
#pragma unroll
    for (int r8 = 0; r8 < 8; ++r8) {
      int m = mrb + r8;
      float d2 = qn[m] + snn - 2.f * acc[r8];
      float v = expf(-sqrtf(fmaxf(d2, 0.f)));
      if (qx0[m] == s0 && qx1[m] == s1) v += bcv;
      yp[r8] = fmaf(v, svn, yp[r8]);
    }
  }
#pragma unroll
  for (int r8 = 0; r8 < 8; ++r8) atomicAdd(&yacc[mrb + r8], yp[r8]);
  __syncthreads();
  if (t < 32) out[q0 + t] = qm[t] + yacc[t];
}

// ---------------------------------------------------------------------------
extern "C" void kernel_launch(void* const* d_in, const int* in_sizes, int n_in,
                              void* d_out, int out_size, void* d_ws, size_t ws_size,
                              hipStream_t stream) {
  (void)in_sizes; (void)n_in; (void)out_size; (void)ws_size;
  const float* support = (const float*)d_in[0];
  const float* query   = (const float*)d_in[1];
  const float* WZ1 = (const float*)d_in[2];
  const float* bZ1 = (const float*)d_in[3];
  const float* WZ2 = (const float*)d_in[4];
  const float* bZ2 = (const float*)d_in[5];
  const float* WB1 = (const float*)d_in[6];
  const float* bB1 = (const float*)d_in[7];
  const float* WB2 = (const float*)d_in[8];
  const float* bB2 = (const float*)d_in[9];
  const float* WM1 = (const float*)d_in[10];
  const float* bM1 = (const float*)d_in[11];
  const float* WM2 = (const float*)d_in[12];
  const float* bM2 = (const float*)d_in[13];
  const float* WK1 = (const float*)d_in[14];
  const float* bK1 = (const float*)d_in[15];
  const float* WK2 = (const float*)d_in[16];
  const float* bK2 = (const float*)d_in[17];
  float* out = (float*)d_out;

  // workspace layout (all 16B-aligned; total ~1.72 MB)
  char* ws = (char*)d_ws;
  float* h1mean        = (float*)(ws + 0);
  float* Z             = (float*)(ws + 1024);
  float* cM            = (float*)(ws + 2048);
  float* cK            = (float*)(ws + 3072);
  float* bscal         = (float*)(ws + 4096);
  float* sn            = (float*)(ws + 4352);
  float* rhs           = (float*)(ws + 6400);
  float* sv            = (float*)(ws + 8448);
  unsigned short* WK2p = (unsigned short*)(ws + 12288);                       // 128 KB
  unsigned short* KsRM = (unsigned short*)(ws + 12288 + 131072);              // 256 KB
  unsigned short* KsP  = (unsigned short*)(ws + 12288 + 131072 + 262144);     // 256 KB
  float* selfk         = (float*)(ws + 12288 + 131072 + 262144 + 262144);     // 1 MB

  k_embed<<<1, 256, 0, stream>>>(support, WZ1, bZ1, h1mean);
  k_z<<<1, 256, 0, stream>>>(h1mean, WZ2, bZ2, Z);
  k_context<<<1, 256, 0, stream>>>(Z, WM1, bM1, WK1, bK1, WB1, bB1, WB2, bB2, cM, cK, bscal);
  k_pack_wk2<<<256, 256, 0, stream>>>(WK2, WK2p, sn);
  k_support<<<16, 256, 0, stream>>>(support, WM1, WM2, bM2, WK1, bK2, cM, cK,
                                    WK2p, KsRM, KsP, sn, rhs);
  k_selfk<<<128, 256, 0, stream>>>(KsRM, KsP, sn, support, bscal, selfk);
  k_cg<<<1, 1024, 0, stream>>>(selfk, rhs, sv);
  k_query<<<NQ / 32, 256, 0, stream>>>(query, support, WM1, WM2, bM2, WK1, bK2,
                                       cM, cK, WK2p, KsP, sn, sv, bscal, out);
}
